// CrossAttentionFusion_35287451304359
// MI455X (gfx1250) — matrix operand order repaired and checked by reading, not compiled, not run on hardware
//
#include <hip/hip_runtime.h>
#include <hip/hip_bf16.h>

#define BB      32
#define HHN     512
#define NMODS   4
#define DD      256
#define MM      16
#define NHEADS  8
#define DHEAD   32
#define FFN_DIM 1024
#define SX      264      // padded bf16/f32 row stride (16B aligned, bank-spread)
#define SHC     72       // FFN hidden-chunk row stride

typedef __attribute__((ext_vector_type(16))) __bf16 v16bf;
typedef __attribute__((ext_vector_type(8)))  float  v8f;

union ABFrag { v16bf v; unsigned int u[8]; };
union CFrag  { v8f  v; float f[8]; };

__device__ __forceinline__ unsigned short f2bf(float x){
  union { float f; unsigned int u; } c; c.f = x;
  unsigned int r = c.u + (0x7FFFu + ((c.u >> 16) & 1u));
  return (unsigned short)(r >> 16);
}
__device__ __forceinline__ float b2f(unsigned short x){
  union { unsigned int u; float f; } c; c.u = ((unsigned int)x) << 16; return c.f;
}

__device__ __forceinline__ v8f wmma_bf16(v16bf a, v16bf b, v8f c){
  return __builtin_amdgcn_wmma_f32_16x16x32_bf16(false, a, false, b, (short)0, c, false, false);
}

// A fragment: 16x32 bf16 tile, row-major source.
// lanes 0-15: M=r, K=k0+0..15; lanes 16-31: M=r, K=k0+16..31
__device__ __forceinline__ v16bf load_fragA(const unsigned short* base, int stride, int k0, int r, int half){
  ABFrag fb;
  const unsigned int* p0 = (const unsigned int*)(base + (size_t)r*stride + k0 + half*16);
  #pragma unroll
  for (int i=0;i<8;++i){ fb.u[i] = p0[i]; }
  return fb.v;
}

// B fragment: 32x16 bf16; source holds B^T row-major (row n = output column n).
// lanes 0-15: col=r, K=k0+0..15; lanes 16-31: col=r, K=k0+16..31
__device__ __forceinline__ v16bf load_fragB(const unsigned short* base, int stride, int k0, int r, int half){
  ABFrag fb;
  const unsigned int* p = (const unsigned int*)(base + (size_t)r*stride + k0 + half*16);
  #pragma unroll
  for (int i=0;i<8;++i) fb.u[i] = p[i];
  return fb.v;
}

// A fragment for attention probs: 16x32, K=16..31 implicit zeros. Row stride = 16 elems.
__device__ __forceinline__ v16bf load_fragA_probs(const unsigned short* base, int r, int half){
  ABFrag fb;
  const unsigned int* p = (const unsigned int*)(base + r*16);
  #pragma unroll
  for (int i=0;i<8;++i){ fb.u[i] = half ? 0u : p[i]; }
  return fb.v;
}

// Single-A GEMM (packed CA K/V projection)
template<int TPW, int K>
__device__ __forceinline__ void gemm_tiles(const unsigned short* A, int astride,
                                           const unsigned short* __restrict__ W,
                                           CFrag* c, int wave, int r, int half){
  #pragma unroll
  for (int t=0;t<TPW;++t)
    #pragma unroll
    for (int i=0;i<8;++i) c[t].f[i] = 0.f;
  for (int k0=0;k0<K;k0+=32){
    v16bf a = load_fragA(A, astride, k0, r, half);
    #pragma unroll
    for (int t=0;t<TPW;++t){
      v16bf b = load_fragB(W + (size_t)(wave*TPW + t)*16*K, K, k0, r, half);
      c[t].v = wmma_bf16(a, b, c[t].v);
    }
  }
}

// Dual-sample GEMM: one weight (B) fragment feeds two A fragments -> halves L2 weight traffic.
template<int TPW, int K>
__device__ __forceinline__ void gemm_tiles2(const unsigned short* A0, const unsigned short* A1,
                                            int astride, const unsigned short* __restrict__ W,
                                            CFrag (*c)[TPW], int wave, int r, int half){
  #pragma unroll
  for (int s=0;s<2;++s)
    #pragma unroll
    for (int t=0;t<TPW;++t)
      #pragma unroll
      for (int i=0;i<8;++i) c[s][t].f[i] = 0.f;
  for (int k0=0;k0<K;k0+=32){
    v16bf a0 = load_fragA(A0, astride, k0, r, half);
    v16bf a1 = load_fragA(A1, astride, k0, r, half);
    #pragma unroll
    for (int t=0;t<TPW;++t){
      v16bf b = load_fragB(W + (size_t)(wave*TPW + t)*16*K, K, k0, r, half);
      c[0][t].v = wmma_bf16(a0, b, c[0][t].v);
      c[1][t].v = wmma_bf16(a1, b, c[1][t].v);
    }
  }
}

// ---- LDS pool: phase-overlaid scratch + persistent state. 132,224 B total ----
// Phase A (cross-attn), inside scratch [0 .. 79,360):
#define A_GXP 0        // 16x264 : packed gated bf16 (rows 0-7 = 2 samples x 4 modalities)
#define A_KXP 8448     // 16x264 : packed K (rows 0-7 valid)
#define A_VTP 16896    // 256x8  : packed V^T (col 4s+j = sample s, key j)
#define A_SC  20992    // 8x16x16 f32 scores (cols 0-7 = packed keys)
#define A_PA  29184    // 8x16x16 bf16 block-diag probs
#define A_CO  33280    // 2 x 16x264 attn out
// Phase B (self-attn), inside scratch:
#define B_Q   0        // 2 x 16x264
#define B_K   16896    // 2 x 16x264
#define B_VT  33792    // 2 x 256x16
#define B_SC  50176    // 8x16x16 f32 (per-sample, reused)
#define B_PA  58368    // 8x16x16 bf16
#define B_CO  62464    // 2 x 16x264
// Phase C (FFN), inside scratch:
#define C_HCH 0        // 2 x 16x72 bf16 hidden chunk (64 cols + pad)
// Persistent:
#define OFF_XS  79360  // 2 x 16x264 f32 : pre-LN accumulation
#define OFF_XH  113152 // 2 x 16x264 bf16: LN output / GEMM A / residual
#define OFF_RED 130048 // 2 x 256 f32
#define OFF_MV  132096 // 16x2 f32
#define POOLSZ  132224

__device__ __forceinline__ void softmax_rows(float* Sc, unsigned short* Pa, int valid, int tid){
  if (tid < 128){
    int h = tid >> 4, m = tid & 15;
    float* srow = Sc + (h*16 + m)*16;
    unsigned short* prow = Pa + (h*16 + m)*16;
    float mx = -1e30f;
    for (int j=0;j<valid;++j) mx = fmaxf(mx, srow[j]);
    float e[16]; float sum = 0.f;
    for (int j=0;j<valid;++j){ e[j] = __expf(srow[j]-mx); sum += e[j]; }
    float inv = 1.f / sum;
    for (int j=0;j<16;++j) prow[j] = (j < valid) ? f2bf(e[j]*inv) : (unsigned short)0;
  }
  __syncthreads();
}

__device__ __forceinline__ void layernorm_16x256(float* Xf, const float* gam, const float* bet,
    float* Red1, float* Red2, float* Mv, unsigned short* Xh, float* gout, int tid){
  int row = tid >> 4, c16 = tid & 15, c0 = c16*16;
  float* xr = Xf + row*SX;
  float s1 = 0.f, s2 = 0.f;
  #pragma unroll
  for (int j=0;j<16;++j){ float v = xr[c0+j]; s1 += v; s2 += v*v; }
  Red1[row*16+c16] = s1; Red2[row*16+c16] = s2;
  __syncthreads();
  if (c16 == 0){
    float a = 0.f, b = 0.f;
    #pragma unroll
    for (int j=0;j<16;++j){ a += Red1[row*16+j]; b += Red2[row*16+j]; }
    float mean = a * (1.f/256.f);
    float var  = b * (1.f/256.f) - mean*mean;
    Mv[row*2]   = mean;
    Mv[row*2+1] = rsqrtf(var + 1e-5f);
  }
  __syncthreads();
  float mean = Mv[row*2], rstd = Mv[row*2+1];
  for (int j=0;j<16;++j){
    int c = c0+j;
    float y = (xr[c]-mean)*rstd*gam[c] + bet[c];
    if (gout) gout[row*256 + c] = y;
    else      Xh[row*SX + c] = f2bf(y);
  }
  __syncthreads();
}

// ---- prep kernels ----
__global__ void cvt_bf16(const float* __restrict__ src, unsigned short* __restrict__ dst, int n){
  for (int i = blockIdx.x*blockDim.x + threadIdx.x; i < n; i += gridDim.x*blockDim.x)
    dst[i] = f2bf(src[i]);
}

// qp[m][n] = query_tokens[m] . wq[n] + bq[n]  (identical for all 16384 samples -> hoisted)
__global__ void qp_precompute(const float* __restrict__ qt, const float* __restrict__ w,
                              const float* __restrict__ b, unsigned short* __restrict__ qp){
  int n = threadIdx.x;
  for (int m = 0; m < MM; ++m){
    float acc = b[n];
    for (int k = 0; k < DD; ++k) acc += qt[m*DD+k] * w[n*DD+k];
    qp[m*DD+n] = f2bf(acc);
  }
}

// ---- main fused kernel: one workgroup per PAIR of (b,h) samples ----
__global__ __launch_bounds__(256) void fusion_main(
    const float* __restrict__ gated, const float* __restrict__ qt,
    const float* __restrict__ ca_in_b, const float* __restrict__ ca_out_b,
    const float* __restrict__ n1g, const float* __restrict__ n1b,
    const float* __restrict__ sa_in_b, const float* __restrict__ sa_out_b,
    const float* __restrict__ n2g, const float* __restrict__ n2b,
    const float* __restrict__ fb1, const float* __restrict__ fb2,
    const float* __restrict__ n3g, const float* __restrict__ n3b,
    const unsigned short* __restrict__ kvw, const unsigned short* __restrict__ cow,
    const unsigned short* __restrict__ saw, const unsigned short* __restrict__ sow,
    const unsigned short* __restrict__ f1w, const unsigned short* __restrict__ f2w,
    const unsigned short* __restrict__ qp, float* __restrict__ out)
{
  __shared__ __align__(16) unsigned char pool[POOLSZ];
  unsigned char* scr = pool;
  float*          XS   = (float*)(pool + OFF_XS);          // [2][16][SX]
  unsigned short* XH   = (unsigned short*)(pool + OFF_XH); // [2][16][SX]
  float*          Red1 = (float*)(pool + OFF_RED);
  float*          Red2 = Red1 + 256;
  float*          Mv   = (float*)(pool + OFF_MV);

  const int tid  = threadIdx.x;
  const int wave = tid >> 5, lane = tid & 31, half = lane >> 4, r = lane & 15;
  const size_t sbase = (size_t)blockIdx.x * 2;
  const float* gsrc[2] = { gated + (sbase  )*(NMODS*DD), gated + (sbase+1)*(NMODS*DD) };
  float*       osrc[2] = { out   + (sbase  )*(MM*DD),    out   + (sbase+1)*(MM*DD) };
  const float  scale = 0.17677669529663687f; // 1/sqrt(32)

  // ================= Phase A: cross-attention =================
  unsigned short* GXP = (unsigned short*)(scr + A_GXP);
  unsigned short* KXP = (unsigned short*)(scr + A_KXP);
  unsigned short* VTP = (unsigned short*)(scr + A_VTP);
  float*          ASC = (float*)(scr + A_SC);
  unsigned short* APA = (unsigned short*)(scr + A_PA);
  unsigned short* ACO = (unsigned short*)(scr + A_CO);

  // stage 0: pack both samples' gated (2x4 rows) into one 16-row A matrix
  for (int idx = tid; idx < MM*SX; idx += 256){
    int row = idx / SX, col = idx - row*SX;
    unsigned short v = 0;
    if (row < 8 && col < DD) v = f2bf(gsrc[row>>2][(row&3)*DD + col]);
    GXP[idx] = v;
  }
  __syncthreads();

  // stage 1: packed K/V projection (one GEMM covers both samples)
  {
    CFrag c[4];
    gemm_tiles<4,DD>(GXP, SX, kvw, c, wave, r, half);
    #pragma unroll
    for (int t=0;t<4;++t){
      int tile = wave*4 + t;
      if (tile < 16){
        int n = tile*16 + r; float bias = ca_in_b[DD + n];
        #pragma unroll
        for (int i=0;i<8;++i){ int m = i + half*8; KXP[m*SX+n] = f2bf(c[t].f[i] + bias); }
      } else {
        int n = (tile-16)*16 + r; float bias = ca_in_b[2*DD + n];
        #pragma unroll
        for (int i=0;i<8;++i){ int m = i + half*8; if (m < 8) VTP[n*8+m] = f2bf(c[t].f[i] + bias); }
      }
    }
  }
  __syncthreads();

  // stage 2: CA scores — one WMMA per head covers BOTH samples (shared queries, packed keys)
  {
    CFrag c;
    #pragma unroll
    for (int i=0;i<8;++i) c.f[i]=0.f;
    v16bf a = load_fragA(qp, DD, DHEAD*wave, r, half);
    v16bf b = load_fragB(KXP, SX, DHEAD*wave, r, half);
    c.v = wmma_bf16(a, b, c.v);
    #pragma unroll
    for (int i=0;i<8;++i){ int m = i + half*8; ASC[(wave*16+m)*16 + r] = c.f[i]*scale; }
  }
  __syncthreads();

  // stage 3: per-sample masked softmax over packed key columns
  if (tid < 128){
    int h = tid >> 4, m = tid & 15;
    float* srow = ASC + (h*16+m)*16;
    unsigned short* prow = APA + (h*16+m)*16;
    for (int s=0;s<2;++s){
      int jb = s*4;
      float mx = -1e30f;
      for (int j=0;j<NMODS;++j) mx = fmaxf(mx, srow[jb+j]);
      float e[NMODS]; float sum = 0.f;
      for (int j=0;j<NMODS;++j){ e[j] = __expf(srow[jb+j]-mx); sum += e[j]; }
      float inv = 1.f / sum;
      for (int j=0;j<NMODS;++j) prow[jb+j] = f2bf(e[j]*inv);
    }
    for (int j=8;j<16;++j) prow[j] = 0;
  }
  __syncthreads();

  // stage 4: attn @ V with block-diagonal B (cross-sample terms vanish exactly)
  {
    v16bf a = load_fragA_probs(APA + wave*256, r, half);
    #pragma unroll
    for (int t=0;t<4;++t){
      int s = t >> 1, dt = t & 1;
      int dcol = DHEAD*wave + dt*16 + r;
      ABFrag fb;
      #pragma unroll
      for (int i=0;i<8;++i) fb.u[i] = 0u;
      if (half == 0){
        const unsigned int* p = (const unsigned int*)(VTP + dcol*8 + s*4);
        fb.u[2*s] = p[0]; fb.u[2*s+1] = p[1];
      }
      CFrag c;
      #pragma unroll
      for (int i=0;i<8;++i) c.f[i]=0.f;
      c.v = wmma_bf16(a, fb.v, c.v);
      unsigned short* Co_s = ACO + s*(MM*SX);
      #pragma unroll
      for (int i=0;i<8;++i){ int m = i + half*8; Co_s[m*SX+dcol] = f2bf(c.f[i]); }
    }
  }
  __syncthreads();

  // stage 5: CA out-proj, dual-sample (weight fragment reused x2) + query residual
  {
    CFrag c[2][2];
    gemm_tiles2<2,DD>(ACO, ACO + MM*SX, SX, cow, c, wave, r, half);
    #pragma unroll
    for (int t=0;t<2;++t){
      int n = (wave*2+t)*16 + r; float bias = ca_out_b[n];
      #pragma unroll
      for (int s=0;s<2;++s){
        float* Xs = XS + s*(MM*SX);
        #pragma unroll
        for (int i=0;i<8;++i){ int m = i + half*8; Xs[m*SX+n] = c[s][t].f[i] + bias + qt[m*DD+n]; }
      }
    }
  }
  __syncthreads();

  // stage 6: LN1 per sample
  for (int s=0;s<2;++s)
    layernorm_16x256(XS + s*MM*SX, n1g, n1b, Red1, Red2, Mv, XH + s*MM*SX, nullptr, tid);

  // ================= Phase B: self-attention =================
  unsigned short* BQ  = (unsigned short*)(scr + B_Q);
  unsigned short* BK  = (unsigned short*)(scr + B_K);
  unsigned short* BVT = (unsigned short*)(scr + B_VT);
  float*          BSC = (float*)(scr + B_SC);
  unsigned short* BPA = (unsigned short*)(scr + B_PA);
  unsigned short* BCO = (unsigned short*)(scr + B_CO);

  // stage 7: SA QKV projection, dual-sample
  {
    CFrag c[2][6];
    gemm_tiles2<6,DD>(XH, XH + MM*SX, SX, saw, c, wave, r, half);
    #pragma unroll
    for (int t=0;t<6;++t){
      int tile = wave*6 + t;
      float bias = sa_in_b[tile*16 + r];
      #pragma unroll
      for (int s=0;s<2;++s){
        if (tile < 16){
          int n = tile*16 + r;
          unsigned short* Qd = BQ + s*MM*SX;
          #pragma unroll
          for (int i=0;i<8;++i){ int m = i + half*8; Qd[m*SX+n] = f2bf(c[s][t].f[i] + bias); }
        } else if (tile < 32){
          int n = (tile-16)*16 + r;
          unsigned short* Kd = BK + s*MM*SX;
          #pragma unroll
          for (int i=0;i<8;++i){ int m = i + half*8; Kd[m*SX+n] = f2bf(c[s][t].f[i] + bias); }
        } else {
          int n = (tile-32)*16 + r;
          unsigned short* Vd = BVT + s*(DD*16);
          #pragma unroll
          for (int i=0;i<8;++i){ int m = i + half*8; Vd[n*16+m] = f2bf(c[s][t].f[i] + bias); }
        }
      }
    }
  }
  __syncthreads();

  // stages 8-10: attention core, sample-sequential (reuses SC/PA scratch)
  for (int s=0;s<2;++s){
    const unsigned short* Qs = BQ + s*MM*SX;
    const unsigned short* Ks = BK + s*MM*SX;
    const unsigned short* Vs = BVT + s*(DD*16);
    {
      CFrag c;
      #pragma unroll
      for (int i=0;i<8;++i) c.f[i]=0.f;
      v16bf a = load_fragA(Qs, SX, DHEAD*wave, r, half);
      v16bf b = load_fragB(Ks, SX, DHEAD*wave, r, half);
      c.v = wmma_bf16(a, b, c.v);
      #pragma unroll
      for (int i=0;i<8;++i){ int m = i + half*8; BSC[(wave*16+m)*16 + r] = c.f[i]*scale; }
    }
    __syncthreads();
    softmax_rows(BSC, BPA, 16, tid);
    {
      v16bf a = load_fragA_probs(BPA + wave*256, r, half);
      #pragma unroll
      for (int t=0;t<2;++t){
        CFrag c;
        #pragma unroll
        for (int i=0;i<8;++i) c.f[i]=0.f;
        ABFrag fb;
        if (half == 0){
          const unsigned int* p = (const unsigned int*)(Vs + (DHEAD*wave + t*16 + r)*16);
          #pragma unroll
          for (int i=0;i<8;++i) fb.u[i] = p[i];
        } else {
          #pragma unroll
          for (int i=0;i<8;++i) fb.u[i] = 0u;
        }
        c.v = wmma_bf16(a, fb.v, c.v);
        int n = DHEAD*wave + t*16 + r;
        unsigned short* Co_s = BCO + s*MM*SX;
        #pragma unroll
        for (int i=0;i<8;++i){ int m = i + half*8; Co_s[m*SX+n] = f2bf(c.f[i]); }
      }
    }
    __syncthreads();
  }

  // stage 11: SA out-proj, dual-sample + residual (residual = LN1 output in XH)
  {
    CFrag c[2][2];
    gemm_tiles2<2,DD>(BCO, BCO + MM*SX, SX, sow, c, wave, r, half);
    #pragma unroll
    for (int t=0;t<2;++t){
      int n = (wave*2+t)*16 + r; float bias = sa_out_b[n];
      #pragma unroll
      for (int s=0;s<2;++s){
        float* Xs = XS + s*MM*SX;
        const unsigned short* Xh = XH + s*MM*SX;
        #pragma unroll
        for (int i=0;i<8;++i){ int m = i + half*8; Xs[m*SX+n] = b2f(Xh[m*SX+n]) + c[s][t].f[i] + bias; }
      }
    }
  }
  __syncthreads();

  // stage 12: LN2 per sample
  for (int s=0;s<2;++s)
    layernorm_16x256(XS + s*MM*SX, n2g, n2b, Red1, Red2, Mv, XH + s*MM*SX, nullptr, tid);

  // ================= Phase C: FFN, hidden-chunked (no 32KB hidden buffer) =================
  unsigned short* HCH = (unsigned short*)(scr + C_HCH); // [2][16][SHC]
  {
    CFrag c2[2][2]; // FFN2 accumulators live across all chunks
    #pragma unroll
    for (int s=0;s<2;++s)
      #pragma unroll
      for (int t=0;t<2;++t)
        #pragma unroll
        for (int i=0;i<8;++i) c2[s][t].f[i] = 0.f;

    for (int hc = 0; hc < FFN_DIM/64; ++hc){
      // FFN1 chunk: each wave computes one 16x16 hidden tile (wave -> sample, col-tile)
      {
        int s = wave >> 2, ct = wave & 3;
        const unsigned short* Xh = XH + s*MM*SX;
        int ncol = hc*64 + ct*16 + r;
        CFrag c;
        #pragma unroll
        for (int i=0;i<8;++i) c.f[i]=0.f;
        for (int k0=0;k0<DD;k0+=32){
          v16bf a = load_fragA(Xh, SX, k0, r, half);
          v16bf b = load_fragB(f1w + (size_t)(hc*64 + ct*16)*DD, DD, k0, r, half);
          c.v = wmma_bf16(a, b, c.v);
        }
        float bias = fb1[ncol];
        unsigned short* Hs = HCH + s*(MM*SHC);
        #pragma unroll
        for (int i=0;i<8;++i){
          int m = i + half*8;
          float v = c.f[i] + bias;
          v = 0.5f * v * (1.f + erff(v * 0.70710678118654752f));
          Hs[m*SHC + ct*16 + r] = f2bf(v);
        }
      }
      __syncthreads();
      // FFN2 partial accumulation over this 64-wide hidden chunk
      #pragma unroll
      for (int s=0;s<2;++s){
        const unsigned short* Hs = HCH + s*(MM*SHC);
        #pragma unroll
        for (int k0=0;k0<64;k0+=32){
          v16bf a = load_fragA(Hs, SHC, k0, r, half);
          #pragma unroll
          for (int t=0;t<2;++t){
            v16bf b = load_fragB(f2w + (size_t)(wave*2+t)*16*FFN_DIM, FFN_DIM, hc*64 + k0, r, half);
            c2[s][t].v = wmma_bf16(a, b, c2[s][t].v);
          }
        }
      }
      __syncthreads();
    }

    // FFN epilogue: bias + residual (LN2 output in XH)
    #pragma unroll
    for (int t=0;t<2;++t){
      int n = (wave*2+t)*16 + r; float bias = fb2[n];
      #pragma unroll
      for (int s=0;s<2;++s){
        float* Xs = XS + s*MM*SX;
        const unsigned short* Xh = XH + s*MM*SX;
        #pragma unroll
        for (int i=0;i<8;++i){ int m = i + half*8; Xs[m*SX+n] = b2f(Xh[m*SX+n]) + c2[s][t].f[i] + bias; }
      }
    }
  }
  __syncthreads();

  // stage 15: LN3 per sample -> global output
  for (int s=0;s<2;++s)
    layernorm_16x256(XS + s*MM*SX, n3g, n3b, Red1, Red2, Mv, XH + s*MM*SX, osrc[s], tid);
}

// ---- workspace layout (bf16 elements) ----
#define WS_KVW 0         // 512*256
#define WS_COW 131072    // 256*256
#define WS_SAW 196608    // 768*256
#define WS_SOW 393216    // 256*256
#define WS_F1W 458752    // 1024*256
#define WS_F2W 720896    // 256*1024
#define WS_QP  983040    // 16*256

extern "C" void kernel_launch(void* const* d_in, const int* in_sizes, int n_in,
                              void* d_out, int out_size, void* d_ws, size_t ws_size,
                              hipStream_t stream){
  (void)in_sizes; (void)n_in; (void)out_size; (void)ws_size;
  const float* gated    = (const float*)d_in[0];
  const float* qt       = (const float*)d_in[1];
  const float* ca_in_w  = (const float*)d_in[2];
  const float* ca_in_b  = (const float*)d_in[3];
  const float* ca_out_w = (const float*)d_in[4];
  const float* ca_out_b = (const float*)d_in[5];
  const float* n1g      = (const float*)d_in[6];
  const float* n1b      = (const float*)d_in[7];
  const float* sa_in_w  = (const float*)d_in[8];
  const float* sa_in_b  = (const float*)d_in[9];
  const float* sa_out_w = (const float*)d_in[10];
  const float* sa_out_b = (const float*)d_in[11];
  const float* n2g      = (const float*)d_in[12];
  const float* n2b      = (const float*)d_in[13];
  const float* ffn_w1   = (const float*)d_in[14];
  const float* ffn_b1   = (const float*)d_in[15];
  const float* ffn_w2   = (const float*)d_in[16];
  const float* ffn_b2   = (const float*)d_in[17];
  const float* n3g      = (const float*)d_in[18];
  const float* n3b      = (const float*)d_in[19];

  unsigned short* ws  = (unsigned short*)d_ws;
  unsigned short* kvw = ws + WS_KVW;
  unsigned short* cow = ws + WS_COW;
  unsigned short* saw = ws + WS_SAW;
  unsigned short* sow = ws + WS_SOW;
  unsigned short* f1w = ws + WS_F1W;
  unsigned short* f2w = ws + WS_F2W;
  unsigned short* qp  = ws + WS_QP;

  cvt_bf16<<<256, 256, 0, stream>>>(ca_in_w + DD*DD, kvw, 2*DD*DD);   // wk,wv
  cvt_bf16<<<256, 256, 0, stream>>>(ca_out_w,        cow, DD*DD);
  cvt_bf16<<<256, 256, 0, stream>>>(sa_in_w,         saw, 3*DD*DD);
  cvt_bf16<<<256, 256, 0, stream>>>(sa_out_w,        sow, DD*DD);
  cvt_bf16<<<512, 256, 0, stream>>>(ffn_w1,          f1w, FFN_DIM*DD);
  cvt_bf16<<<512, 256, 0, stream>>>(ffn_w2,          f2w, DD*FFN_DIM);
  qp_precompute<<<1, 256, 0, stream>>>(qt, ca_in_w, ca_in_b, qp);

  fusion_main<<<(BB*HHN)/2, 256, 0, stream>>>(gated, qt, ca_in_b, ca_out_b,
      n1g, n1b, sa_in_b, sa_out_b, n2g, n2b, ffn_b1, ffn_b2, n3g, n3b,
      kvw, cow, saw, sow, f1w, f2w, qp, (float*)d_out);
}
